// Encoder_81707457839724
// MI455X (gfx1250) — compile-verified, matrix operand order checked by reading, not run on hardware
//
#include <hip/hip_runtime.h>
#include <hip/hip_bf16.h>

// ---------------- problem constants (match reference) ----------------
#define NNODES 100000
#define IN_C   512
#define HID    256
#define OUT_C  256
#define EPSBN  1e-5f

typedef __attribute__((ext_vector_type(16))) __bf16 v16bf;
typedef __attribute__((ext_vector_type(8)))  __bf16 v8bf;
typedef __attribute__((ext_vector_type(8)))  float  v8f;
typedef __attribute__((ext_vector_type(4)))  float  f4;

// ---------------------------------------------------------------------
// Zero-fill (graph-capture-safe replacement for hipMemsetAsync)
// ---------------------------------------------------------------------
__global__ void zero_f32(float* __restrict__ p, size_t n) {
    size_t i = (size_t)blockIdx.x * blockDim.x + threadIdx.x;
    size_t stride = (size_t)gridDim.x * blockDim.x;
    for (; i < n; i += stride) p[i] = 0.0f;
}

// ---------------------------------------------------------------------
// Split W[K x Nout] fp32 into transposed bf16 hi/lo: Wt[Nout x K]
// (row-major over Nout so each lane's B-fragment k-run is contiguous)
// ---------------------------------------------------------------------
__global__ void split_wt(const float* __restrict__ W,
                         __bf16* __restrict__ Wt_hi,
                         __bf16* __restrict__ Wt_lo,
                         int K, int Nout) {
    int idx = blockIdx.x * blockDim.x + threadIdx.x;
    if (idx >= K * Nout) return;
    int k = idx / Nout, n = idx % Nout;
    float v = W[idx];
    __bf16 h = (__bf16)v;
    float  r = v - (float)h;
    Wt_hi[(size_t)n * K + k] = h;
    Wt_lo[(size_t)n * K + k] = (__bf16)r;
}

// ---------------------------------------------------------------------
// GEMM: C[M x Nout] = f(A)[M x K] @ W[K x Nout] (+ bias_out)
// split-bf16 x3 WMMA accumulation in f32.
//   MODE 0: f(a) = a
//   MODE 1: f(a) = a + bias_in[k]                 (fold b_gcn)
//   MODE 2: f(a) = relu(a*scale_in[k]+shift_in[k]) (fold BN1+ReLU)
// Block: 256 thr = 8 waves. Wave = 16(M) x 64(N). Block = 32(M) x 256(N).
// Requires M % 32 == 0, K % 32 == 0, Nout == 256.
// ---------------------------------------------------------------------
template <int MODE>
__global__ __launch_bounds__(256) void gemm_bf16x3(
    const float*  __restrict__ A,
    const __bf16* __restrict__ Bt_hi,   // [Nout x K]
    const __bf16* __restrict__ Bt_lo,   // [Nout x K]
    float*        __restrict__ C,
    const float*  __restrict__ bias_in,
    const float*  __restrict__ scale_in,
    const float*  __restrict__ shift_in,
    const float*  __restrict__ bias_out,
    int M, int K, int Nout)
{
    const int lane  = threadIdx.x & 31;
    const int wave  = threadIdx.x >> 5;    // wave32
    const int waveM = wave >> 2;           // 0..1
    const int waveN = wave & 3;            // 0..3
    const int row0  = blockIdx.x * 32 + waveM * 16;
    const int col0  = waveN * 64;
    const int lm    = lane & 15;           // A-row / B-col within tile
    const int lh    = lane >> 4;           // k-half selector

    const float* __restrict__ arow = A + (size_t)(row0 + lm) * K;

    v8f acc[4];
    #pragma unroll
    for (int t = 0; t < 4; ++t) acc[t] = (v8f){};

    for (int kb = 0; kb < K; kb += 32) {
        const int k0 = kb + lh * 8;
        // ---- A fragment: 16 floats (K = k0..k0+7, k0+16..k0+23) ----
        f4 a0 = *(const f4*)(arow + k0);
        f4 a1 = *(const f4*)(arow + k0 + 4);
        f4 a2 = *(const f4*)(arow + k0 + 16);
        f4 a3 = *(const f4*)(arow + k0 + 20);
        float af[16] = { a0.x, a0.y, a0.z, a0.w,  a1.x, a1.y, a1.z, a1.w,
                         a2.x, a2.y, a2.z, a2.w,  a3.x, a3.y, a3.z, a3.w };
        if constexpr (MODE == 1) {
            #pragma unroll
            for (int j = 0; j < 8; ++j) {
                af[j]     += bias_in[k0 + j];
                af[8 + j] += bias_in[k0 + 16 + j];
            }
        } else if constexpr (MODE == 2) {
            #pragma unroll
            for (int j = 0; j < 8; ++j) {
                float v0 = fmaf(af[j],     scale_in[k0 + j],      shift_in[k0 + j]);
                float v1 = fmaf(af[8 + j], scale_in[k0 + 16 + j], shift_in[k0 + 16 + j]);
                af[j]     = v0 > 0.0f ? v0 : 0.0f;
                af[8 + j] = v1 > 0.0f ? v1 : 0.0f;
            }
        }
        v16bf a_hi, a_lo;
        #pragma unroll
        for (int j = 0; j < 16; ++j) {
            __bf16 h = (__bf16)af[j];
            a_hi[j] = h;
            a_lo[j] = (__bf16)(af[j] - (float)h);
        }
        // ---- 4 B tiles along N, hi/lo each ----
        #pragma unroll
        for (int t = 0; t < 4; ++t) {
            const size_t nbase = (size_t)(col0 + t * 16 + lm) * K + k0;
            v8bf bh0 = *(const v8bf*)(Bt_hi + nbase);
            v8bf bh1 = *(const v8bf*)(Bt_hi + nbase + 16);
            v8bf bl0 = *(const v8bf*)(Bt_lo + nbase);
            v8bf bl1 = *(const v8bf*)(Bt_lo + nbase + 16);
            v16bf b_hi = __builtin_shufflevector(bh0, bh1, 0,1,2,3,4,5,6,7,8,9,10,11,12,13,14,15);
            v16bf b_lo = __builtin_shufflevector(bl0, bl1, 0,1,2,3,4,5,6,7,8,9,10,11,12,13,14,15);
            acc[t] = __builtin_amdgcn_wmma_f32_16x16x32_bf16(false, a_hi, false, b_hi, (short)0, acc[t], false, false);
            acc[t] = __builtin_amdgcn_wmma_f32_16x16x32_bf16(false, a_lo, false, b_hi, (short)0, acc[t], false, false);
            acc[t] = __builtin_amdgcn_wmma_f32_16x16x32_bf16(false, a_hi, false, b_lo, (short)0, acc[t], false, false);
        }
    }

    // ---- epilogue: C/D layout row = i + 8*lh, col = lm ----
    const int orow = row0 + lh * 8;
    #pragma unroll
    for (int t = 0; t < 4; ++t) {
        const int n = col0 + t * 16 + lm;
        const float b = bias_out ? bias_out[n] : 0.0f;
        #pragma unroll
        for (int i = 0; i < 8; ++i)
            C[(size_t)(orow + i) * Nout + n] = acc[t][i] + b;
    }
}

// ---------------------------------------------------------------------
// SpMM scatter: out[row[e]] += val[e] * H[col[e]]  (256 channels)
// One wave per edge; lane covers 2 float4 (8 channels).
// ---------------------------------------------------------------------
__global__ __launch_bounds__(256) void spmm_scatter(
    const float* __restrict__ H,
    const int*   __restrict__ erow,
    const int*   __restrict__ ecol,
    const float* __restrict__ eval,
    float*       __restrict__ out,
    int E)
{
    const int lane  = threadIdx.x & 31;
    int gwarp = (int)((blockIdx.x * blockDim.x + threadIdx.x) >> 5);
    const int nwarp = (int)((gridDim.x * blockDim.x) >> 5);
    for (int e = gwarp; e < E; e += nwarp) {
        const int r = erow[e], c = ecol[e];
        const float v = eval[e];
        const f4* __restrict__ src = (const f4*)(H + (size_t)c * HID);
        float*    __restrict__ dst = out + (size_t)r * HID;
        #pragma unroll
        for (int p = 0; p < 2; ++p) {
            const int q = lane + p * 32;
            f4 d = src[q];
            unsafeAtomicAdd(dst + q * 4 + 0, v * d.x);
            unsafeAtomicAdd(dst + q * 4 + 1, v * d.y);
            unsafeAtomicAdd(dst + q * 4 + 2, v * d.z);
            unsafeAtomicAdd(dst + q * 4 + 3, v * d.w);
        }
    }
}

// ---------------------------------------------------------------------
// BatchNorm: per-channel sum / sumsq over rows (C == 256 == blockDim)
// ---------------------------------------------------------------------
__global__ __launch_bounds__(256) void bn_stats(
    const float* __restrict__ Y, float* __restrict__ sums,
    float* __restrict__ sumsq, int N)
{
    const int c = threadIdx.x;
    float s = 0.0f, sq = 0.0f;
    for (int r = blockIdx.x; r < N; r += gridDim.x) {
        float v = Y[(size_t)r * 256 + c];
        s += v;
        sq = fmaf(v, v, sq);
    }
    unsafeAtomicAdd(&sums[c], s);
    unsafeAtomicAdd(&sumsq[c], sq);
}

__global__ void bn_finalize(const float* __restrict__ sums,
                            const float* __restrict__ sumsq,
                            const float* __restrict__ g,
                            const float* __restrict__ be,
                            float* __restrict__ scale,
                            float* __restrict__ shift, int N)
{
    const int c = threadIdx.x;
    const float inv = 1.0f / (float)N;
    float mean = sums[c] * inv;
    float var  = fmaf(-mean, mean, sumsq[c] * inv);
    float rstd = rsqrtf(var + EPSBN);
    float sc   = g[c] * rstd;
    scale[c] = sc;
    shift[c] = fmaf(-mean, sc, be[c]);
}

__global__ void bn_apply(const float* __restrict__ src, float* __restrict__ dst,
                         const float* __restrict__ scale,
                         const float* __restrict__ shift, size_t n4)
{
    size_t i = (size_t)blockIdx.x * blockDim.x + threadIdx.x;
    size_t stride = (size_t)gridDim.x * blockDim.x;
    for (; i < n4; i += stride) {
        int cb = (int)((i & 63) * 4);   // channel base (256 ch = 64 float4)
        f4 v = ((const f4*)src)[i];
        f4 o;
        o.x = fmaf(v.x, scale[cb + 0], shift[cb + 0]);
        o.y = fmaf(v.y, scale[cb + 1], shift[cb + 1]);
        o.z = fmaf(v.z, scale[cb + 2], shift[cb + 2]);
        o.w = fmaf(v.w, scale[cb + 3], shift[cb + 3]);
        ((f4*)dst)[i] = o;
    }
}

// ---------------------------------------------------------------------
extern "C" void kernel_launch(void* const* d_in, const int* in_sizes, int n_in,
                              void* d_out, int out_size, void* d_ws, size_t ws_size,
                              hipStream_t stream) {
    const float* x     = (const float*)d_in[0];
    const int*   erow  = (const int*)  d_in[1];
    const int*   ecol  = (const int*)  d_in[2];
    const float* eval  = (const float*)d_in[3];
    const float* W_gcn = (const float*)d_in[4];
    const float* b_gcn = (const float*)d_in[5];
    const float* W1    = (const float*)d_in[6];
    const float* b1    = (const float*)d_in[7];
    const float* g1    = (const float*)d_in[8];
    const float* be1   = (const float*)d_in[9];
    const float* W2    = (const float*)d_in[10];
    const float* b2    = (const float*)d_in[11];
    const float* g2    = (const float*)d_in[12];
    const float* be2   = (const float*)d_in[13];
    float* out = (float*)d_out;
    const int E = in_sizes[1];

    // ---- workspace carve-up ----
    char* ws = (char*)d_ws;
    float*  bufA = (float*)ws;                         // N x 256 f32
    size_t  off  = (size_t)NNODES * HID * sizeof(float);
    __bf16* wt1h = (__bf16*)(ws + off); off += (size_t)IN_C * HID * 2;
    __bf16* wt1l = (__bf16*)(ws + off); off += (size_t)IN_C * HID * 2;
    __bf16* wt2h = (__bf16*)(ws + off); off += (size_t)HID * HID * 2;
    __bf16* wt2l = (__bf16*)(ws + off); off += (size_t)HID * HID * 2;
    __bf16* wt3h = (__bf16*)(ws + off); off += (size_t)HID * OUT_C * 2;
    __bf16* wt3l = (__bf16*)(ws + off); off += (size_t)HID * OUT_C * 2;
    float*  sums  = (float*)(ws + off); off += 256 * sizeof(float);
    float*  sumsq = (float*)(ws + off); off += 256 * sizeof(float);
    float*  scale = (float*)(ws + off); off += 256 * sizeof(float);
    float*  shift = (float*)(ws + off); off += 256 * sizeof(float);

    const int gemmGrid = NNODES / 32;   // 3125 (exact)

    // 1) split/transpose weights to bf16 hi/lo
    split_wt<<<(IN_C * HID + 255) / 256, 256, 0, stream>>>(W_gcn, wt1h, wt1l, IN_C, HID);
    split_wt<<<(HID  * HID + 255) / 256, 256, 0, stream>>>(W1,    wt2h, wt2l, HID,  HID);
    split_wt<<<(HID * OUT_C + 255) / 256, 256, 0, stream>>>(W2,   wt3h, wt3l, HID,  OUT_C);

    // 2) h0 = x @ W_gcn  -> d_out  (no bias yet; b_gcn fused into GEMM2 load)
    gemm_bf16x3<0><<<gemmGrid, 256, 0, stream>>>(x, wt1h, wt1l, out,
                                                 nullptr, nullptr, nullptr, nullptr,
                                                 NNODES, IN_C, HID);

    // 3) bufA = 0 ; bufA = segment_sum(val * h0[col]) over rows
    zero_f32<<<2048, 256, 0, stream>>>(bufA, (size_t)NNODES * HID);
    spmm_scatter<<<(E + 7) / 8, 256, 0, stream>>>(out, erow, ecol, eval, bufA, E);

    // 4) y1 = (bufA + b_gcn) @ W1 + b1 -> d_out
    gemm_bf16x3<1><<<gemmGrid, 256, 0, stream>>>(bufA, wt2h, wt2l, out,
                                                 b_gcn, nullptr, nullptr, b1,
                                                 NNODES, HID, HID);

    // 5) BN1 stats on y1 -> scale/shift (g1, be1 folded)
    zero_f32<<<1, 256, 0, stream>>>(sums, 512);   // sums+sumsq contiguous
    bn_stats<<<1024, 256, 0, stream>>>(out, sums, sumsq, NNODES);
    bn_finalize<<<1, 256, 0, stream>>>(sums, sumsq, g1, be1, scale, shift, NNODES);

    // 6) y2 = relu(BN1(y1)) @ W2 + b2 -> bufA   (BN+ReLU fused in A-load)
    gemm_bf16x3<2><<<gemmGrid, 256, 0, stream>>>(out, wt3h, wt3l, bufA,
                                                 nullptr, scale, shift, b2,
                                                 NNODES, HID, OUT_C);

    // 7) BN2 stats on y2 -> final normalize into d_out
    zero_f32<<<1, 256, 0, stream>>>(sums, 512);
    bn_stats<<<1024, 256, 0, stream>>>(bufA, sums, sumsq, NNODES);
    bn_finalize<<<1, 256, 0, stream>>>(sums, sumsq, g2, be2, scale, shift, NNODES);
    bn_apply<<<2048, 256, 0, stream>>>(bufA, out, scale, shift,
                                       (size_t)NNODES * OUT_C / 4);
}
